// Random_Attention_7507602834031
// MI455X (gfx1250) — matrix operand order repaired and checked by reading, not compiled, not run on hardware
//
#include <hip/hip_runtime.h>

// Problem constants (from reference)
#define B_  4
#define S_  2048
#define D_  768
#define H_  12
#define DH_ 64
#define M_  (B_ * S_)   // 8192 rows for the projection GEMMs

typedef __attribute__((ext_vector_type(16))) _Float16 v16h;
typedef __attribute__((ext_vector_type(8)))  _Float16 v8h;
typedef __attribute__((ext_vector_type(8)))  float    v8f;

union V16H { v16h v; v8h h[2]; };

__device__ __forceinline__ v8f wmma_f16(v16h a, v16h b, v8f c) {
  // D = A(16x32 f16) * B(32x16 f16) + C(16x16 f32)
  return __builtin_amdgcn_wmma_f32_16x16x32_f16(
      /*neg_a=*/false, a, /*neg_b=*/false, b,
      /*c_mod=*/(short)0, c, /*reuse_a=*/false, /*reuse_b=*/false);
}

// ---------------------------------------------------------------------------
// f32 -> f16 elementwise convert
// ---------------------------------------------------------------------------
__global__ void cvt_f32_to_f16(const float* __restrict__ in,
                               _Float16* __restrict__ out, int n) {
  int i = blockIdx.x * blockDim.x + threadIdx.x;
  if (i < n) out[i] = (_Float16)in[i];
}

// W[k][n] (f32, row-major KxN) -> Wt[n][k] (f16), so GEMM B-fragments load
// 16 contiguous halves per lane.
__global__ void transpose_w_f16(const float* __restrict__ w,
                                _Float16* __restrict__ wt) {
  int i = blockIdx.x * blockDim.x + threadIdx.x;   // i = n*D_ + k
  if (i >= D_ * D_) return;
  int nn = i / D_, k = i - nn * D_;
  wt[i] = (_Float16)w[k * D_ + nn];
}

// ---------------------------------------------------------------------------
// WMMA GEMM: Y[m,n] = sum_k A[m,k] * W[k,n] + bias[n]
//   A  : f16 [M_, D_]
//   Bt : f16 [D_, D_]  (row n holds W[:,n])
// mode 0: f16 out, head-split   [B,H,S,DH]       (Q, K)
// mode 1: f16 out, head-split-T [B,H,DH,S]       (V transposed)
// mode 2: f32 out, flat         [M_, D_]         (final projection)
// Block = 128 threads (4 waves stacked on M); wave tile = 16(M) x 64(N).
// ---------------------------------------------------------------------------
__global__ __launch_bounds__(128) void gemm_wmma(
    const _Float16* __restrict__ A, const _Float16* __restrict__ Bt,
    const float* __restrict__ bias, void* __restrict__ out, int mode) {
  const int wave = threadIdx.x >> 5;
  const int lane = threadIdx.x & 31;
  const int n = lane & 15;        // fragment column / A row
  const int g = lane >> 4;        // half-wave group
  const int m0 = blockIdx.x * 64 + wave * 16;
  const int n0 = blockIdx.y * 64;

  const v8f zero = {0.f, 0.f, 0.f, 0.f, 0.f, 0.f, 0.f, 0.f};
  v8f acc[4] = {zero, zero, zero, zero};

  const _Float16* arow = A + (size_t)(m0 + n) * D_;

  #pragma unroll 4
  for (int k = 0; k < D_; k += 32) {
    // A fragment: lane m=n holds dims {k+16g..+15} (contiguous)
    V16H af;
    af.h[0] = *(const v8h*)(arow + k + 16 * g);
    af.h[1] = *(const v8h*)(arow + k + 16 * g + 8);
    #pragma unroll
    for (int t = 0; t < 4; ++t) {
      // B fragment: lane col = n0+16t+n, halves j -> k + j + 16g (contiguous)
      const _Float16* brow = Bt + (size_t)(n0 + 16 * t + n) * D_ + k + 16 * g;
      V16H bf;
      bf.h[0] = *(const v8h*)(brow);
      bf.h[1] = *(const v8h*)(brow + 8);
      acc[t] = wmma_f16(af.v, bf.v, acc[t]);
    }
  }

  #pragma unroll
  for (int t = 0; t < 4; ++t) {
    const int col = n0 + 16 * t + n;
    const float bv = bias[col];
    #pragma unroll
    for (int r = 0; r < 8; ++r) {
      const int row = m0 + 8 * g + r;       // D-frag row = r + 8g
      const float val = acc[t][r] + bv;
      if (mode == 2) {
        ((float*)out)[(size_t)row * D_ + col] = val;
      } else {
        const int b = row >> 11, s = row & 2047;   // row = b*2048 + s
        const int h = col >> 6,  d = col & 63;     // col = h*64 + d
        _Float16* o16 = (_Float16*)out;
        if (mode == 0)
          o16[(((size_t)(b * H_ + h)) * S_ + s) * DH_ + d] = (_Float16)val;
        else
          o16[(((size_t)(b * H_ + h)) * DH_ + d) * S_ + s] = (_Float16)val;
      }
    }
  }
}

// ---------------------------------------------------------------------------
// Issue this wave's quarter of the async global->LDS copies for one 32-key
// block: K tile = 32 keys x 64 dims f16 (4 KB, contiguous in global),
// V tile = 64 dims x 32 keys f16 (64 rows of 64 B, global row stride 4 KB).
// 4 x global_load_async_to_lds_b128 per wave (ASYNCcnt += 4).
// ---------------------------------------------------------------------------
__device__ __forceinline__ void issue_kv_async(
    const char* Kbb, const char* Vbb, int kb,
    unsigned kldsbase, unsigned vldsbase, int wave, int lane) {
  // K: wave covers 1024 contiguous bytes (2 x 512B instructions)
  {
    const int o = wave * 1024 + lane * 16;
    const char* g0 = Kbb + (size_t)kb * (DH_ * 2) + o;
    unsigned l0 = kldsbase + (unsigned)o;
    asm volatile("global_load_async_to_lds_b128 %0, %1, off"
                 :: "v"(l0), "v"(g0) : "memory");
    const char* g1 = g0 + 512;
    unsigned l1 = l0 + 512u;
    asm volatile("global_load_async_to_lds_b128 %0, %1, off"
                 :: "v"(l1), "v"(g1) : "memory");
  }
  // V: wave covers 16 dim-rows; each instruction moves 8 rows x 64B
  {
    const int row = wave * 16 + (lane >> 2);      // 8 rows per instruction
    const int cb  = (lane & 3) * 16;              // 16B chunk within 64B row
    const char* g0 = Vbb + (size_t)row * (S_ * 2) + (size_t)kb * 2 + cb;
    unsigned l0 = vldsbase + (unsigned)(row * 64 + cb);
    asm volatile("global_load_async_to_lds_b128 %0, %1, off"
                 :: "v"(l0), "v"(g0) : "memory");
    const char* g1 = g0 + 8 * (S_ * 2);
    unsigned l1 = l0 + 8 * 64u;
    asm volatile("global_load_async_to_lds_b128 %0, %1, off"
                 :: "v"(l1), "v"(g1) : "memory");
  }
}

// ---------------------------------------------------------------------------
// Flash attention (one wave = 16 queries; block = 4 waves = 64 queries).
// Computes S^T = K * Q^T per 16-key tile so the score accumulator lands
// directly in P's A-fragment layout (lane = query) -> zero-shuffle P*V.
// K/V tiles are staged in LDS via async copies (double-buffered), shared by
// all 4 waves. Online softmax; mask byte == 0 -> -1e9; scale 1/sqrt(64).
// ---------------------------------------------------------------------------
__global__ __launch_bounds__(128) void flash_attn(
    const _Float16* __restrict__ Q,            // [B,H,S,DH]
    const _Float16* __restrict__ K,            // [B,H,S,DH]
    const _Float16* __restrict__ Vt,           // [B,H,DH,S]
    const unsigned char* __restrict__ mask,    // [S,S] bool bytes
    _Float16* __restrict__ ctx) {              // [B,S,D]
  __shared__ _Float16 ksm[2][32 * DH_];        // 2 x 4 KB: [key][dim]
  __shared__ _Float16 vsm[2][DH_ * 32];        // 2 x 4 KB: [dim][key]

  const int wave = threadIdx.x >> 5;
  const int lane = threadIdx.x & 31;
  const int n = lane & 15;      // query (for S^T frags) / dim (for O frags)
  const int g = lane >> 4;
  const int b = blockIdx.z, h = blockIdx.y;
  const int q0 = blockIdx.x * 64 + wave * 16;

  const size_t bh = (size_t)(b * H_ + h);
  const _Float16* Qb = Q + (bh * S_ + q0) * DH_;
  const char* Kbb = (const char*)(K + bh * S_ * DH_);
  const char* Vbb = (const char*)(Vt + bh * DH_ * S_);
  const unsigned kbase = (unsigned)(size_t)(void*)&ksm[0][0];
  const unsigned vbase = (unsigned)(size_t)(void*)&vsm[0][0];

  // Q as B-fragments (lane col = query q0+n; dims {8g..+7} and {8g+16..+23})
  const _Float16* qrow = Qb + (size_t)n * DH_;
  V16H qb0, qb1;
  qb0.h[0] = *(const v8h*)(qrow + 8 * g);
  qb0.h[1] = *(const v8h*)(qrow + 8 * g + 16);
  qb1.h[0] = *(const v8h*)(qrow + 32 + 8 * g);
  qb1.h[1] = *(const v8h*)(qrow + 32 + 8 * g + 16);

  const unsigned char* mrow = mask + (size_t)(q0 + n) * S_;

  const v8f zero = {0.f, 0.f, 0.f, 0.f, 0.f, 0.f, 0.f, 0.f};
  v8f o[4] = {zero, zero, zero, zero};   // O frags: lane = dim 16t+n, rows = queries r+8g
  float m_run = -3.0e38f, l_run = 0.f;

  // Preload first K/V tile into buffer 0 (ASYNCcnt = 4 per wave)
  issue_kv_async(Kbb, Vbb, /*kb=*/0, kbase, vbase, wave, lane);

  for (int kb = 0; kb < S_; kb += 32) {
    const int buf = (kb >> 5) & 1;
    // Prefetch next tile into the other buffer, then wait for current tile:
    // async loads complete in order, so asynccnt<=4 means this tile landed.
    if (kb + 32 < S_) {
      issue_kv_async(Kbb, Vbb, kb + 32,
                     kbase + (buf ^ 1) * (32 * DH_ * 2),
                     vbase + (buf ^ 1) * (DH_ * 32 * 2), wave, lane);
      asm volatile("s_wait_asynccnt 0x4" ::: "memory");
    } else {
      asm volatile("s_wait_asynccnt 0x0" ::: "memory");
    }
    __syncthreads();   // all waves' shares of this tile are in LDS

    const _Float16* kls = &ksm[buf][0];
    const _Float16* vls = &vsm[buf][0];

    // ---- scores^T, two 16-key tiles, contraction over DH=64 (2 WMMA each)
    v8f s0 = zero, s1 = zero;
    {
      const _Float16* kr0 = kls + (size_t)n * DH_;          // local keys 0..15
      const _Float16* kr1 = kls + (size_t)(16 + n) * DH_;   // local keys 16..31
      V16H ka, kc;
      ka.h[0] = *(const v8h*)(kr0 + 8 * g);
      ka.h[1] = *(const v8h*)(kr0 + 8 * g + 16);
      kc.h[0] = *(const v8h*)(kr0 + 32 + 8 * g);
      kc.h[1] = *(const v8h*)(kr0 + 32 + 8 * g + 16);
      s0 = wmma_f16(ka.v, qb0.v, s0);
      s0 = wmma_f16(kc.v, qb1.v, s0);
      ka.h[0] = *(const v8h*)(kr1 + 8 * g);
      ka.h[1] = *(const v8h*)(kr1 + 8 * g + 16);
      kc.h[0] = *(const v8h*)(kr1 + 32 + 8 * g);
      kc.h[1] = *(const v8h*)(kr1 + 32 + 8 * g + 16);
      s1 = wmma_f16(ka.v, qb0.v, s1);
      s1 = wmma_f16(kc.v, qb1.v, s1);
    }

    // ---- mask + scale (lane holds query q0+n, keys kb+16t+8g+r : consecutive)
    const unsigned long long mm0 = *(const unsigned long long*)(mrow + kb + 8 * g);
    const unsigned long long mm1 = *(const unsigned long long*)(mrow + kb + 16 + 8 * g);
    float sv0[8], sv1[8];
    float lmax = -3.0e38f;
    #pragma unroll
    for (int r = 0; r < 8; ++r) {
      sv0[r] = ((mm0 >> (8 * r)) & 0xffULL) ? s0[r] * 0.125f : -1.0e9f;
      sv1[r] = ((mm1 >> (8 * r)) & 0xffULL) ? s1[r] * 0.125f : -1.0e9f;
      lmax = fmaxf(lmax, fmaxf(sv0[r], sv1[r]));
    }
    // partner lane (same query, other key half) completes the row reduction
    lmax = fmaxf(lmax, __shfl_xor(lmax, 16));
    const float m_new = fmaxf(m_run, lmax);
    const float alpha = __expf(m_run - m_new);
    float p0[8], p1[8], lsum = 0.f;
    #pragma unroll
    for (int r = 0; r < 8; ++r) {
      p0[r] = __expf(sv0[r] - m_new);
      p1[r] = __expf(sv1[r] - m_new);
      lsum += p0[r] + p1[r];
    }
    lsum += __shfl_xor(lsum, 16);
    l_run = l_run * alpha + lsum;
    m_run = m_new;

    // ---- rescale O accumulators (lane needs alpha of queries 8g+r)
    #pragma unroll
    for (int r = 0; r < 8; ++r) {
      const float ar = __shfl(alpha, 8 * g + r);
      o[0][r] *= ar; o[1][r] *= ar; o[2][r] *= ar; o[3][r] *= ar;
    }

    // ---- P as A-fragment: pure lane-local repack of s^T accumulators
    V16H pa;
    #pragma unroll
    for (int r = 0; r < 8; ++r) {
      pa.v[r]     = (_Float16)p0[r];
      pa.v[r + 8] = (_Float16)p1[r];
    }

    // ---- O += P * V  (LDS V rows are contiguous over keys)
    #pragma unroll
    for (int t = 0; t < 4; ++t) {
      const _Float16* vrow = vls + (size_t)(16 * t + n) * 32 + 8 * g;
      V16H vf;
      vf.h[0] = *(const v8h*)(vrow);
      vf.h[1] = *(const v8h*)(vrow + 16);
      o[t] = wmma_f16(pa.v, vf.v, o[t]);
    }

    __syncthreads();   // all waves done reading before buffer is re-targeted
  }

  // ---- epilogue: ctx[b, q, h*64 + d] = O[q, d] / l[q]
  const float inv_l = 1.0f / l_run;
  #pragma unroll
  for (int r = 0; r < 8; ++r) {
    const float il = __shfl(inv_l, 8 * g + r);
    const int q = q0 + 8 * g + r;
    _Float16* cp = ctx + ((size_t)(b * S_ + q)) * D_ + h * DH_;
    cp[n]      = (_Float16)(o[0][r] * il);
    cp[16 + n] = (_Float16)(o[1][r] * il);
    cp[32 + n] = (_Float16)(o[2][r] * il);
    cp[48 + n] = (_Float16)(o[3][r] * il);
  }
}

// ---------------------------------------------------------------------------
extern "C" void kernel_launch(void* const* d_in, const int* in_sizes, int n_in,
                              void* d_out, int out_size, void* d_ws, size_t ws_size,
                              hipStream_t stream) {
  (void)in_sizes; (void)n_in; (void)out_size; (void)ws_size;
  const float* X  = (const float*)d_in[0];
  const float* Wq = (const float*)d_in[1];
  const float* bq = (const float*)d_in[2];
  const float* Wk = (const float*)d_in[3];
  const float* bk = (const float*)d_in[4];
  const float* Wv = (const float*)d_in[5];
  const float* bv = (const float*)d_in[6];
  const float* Wo = (const float*)d_in[7];
  const float* bo = (const float*)d_in[8];
  const unsigned char* mask = (const unsigned char*)d_in[9];

  char* ws = (char*)d_ws;
  const size_t szAct = (size_t)M_ * D_ * sizeof(_Float16);   // 12.58 MB
  const size_t szW   = (size_t)D_ * D_ * sizeof(_Float16);   // 1.18 MB
  _Float16* Xh  = (_Float16*)(ws);                 // also reused for ctx
  _Float16* Qh  = (_Float16*)(ws + szAct);
  _Float16* Kh  = (_Float16*)(ws + 2 * szAct);
  _Float16* Vt  = (_Float16*)(ws + 3 * szAct);
  _Float16* Wqt = (_Float16*)(ws + 4 * szAct);
  _Float16* Wkt = (_Float16*)(ws + 4 * szAct + szW);
  _Float16* Wvt = (_Float16*)(ws + 4 * szAct + 2 * szW);
  _Float16* Wot = (_Float16*)(ws + 4 * szAct + 3 * szW);
  _Float16* ctxh = Xh;   // Xh is dead after the three QKV GEMMs

  const int nX = M_ * D_;
  cvt_f32_to_f16<<<(nX + 255) / 256, 256, 0, stream>>>(X, Xh, nX);
  const int nW = D_ * D_;
  transpose_w_f16<<<(nW + 255) / 256, 256, 0, stream>>>(Wq, Wqt);
  transpose_w_f16<<<(nW + 255) / 256, 256, 0, stream>>>(Wk, Wkt);
  transpose_w_f16<<<(nW + 255) / 256, 256, 0, stream>>>(Wv, Wvt);
  transpose_w_f16<<<(nW + 255) / 256, 256, 0, stream>>>(Wo, Wot);

  dim3 gg(M_ / 64, D_ / 64, 1);
  gemm_wmma<<<gg, 128, 0, stream>>>(Xh, Wqt, bq, Qh, 0);
  gemm_wmma<<<gg, 128, 0, stream>>>(Xh, Wkt, bk, Kh, 0);
  gemm_wmma<<<gg, 128, 0, stream>>>(Xh, Wvt, bv, Vt, 1);

  dim3 ga(S_ / 64, H_, B_);
  flash_attn<<<ga, 128, 0, stream>>>(Qh, Kh, Vt, mask, ctxh);

  gemm_wmma<<<gg, 128, 0, stream>>>(ctxh, Wot, bo, d_out, 2);
}
